// ForestTrajectoryAE_61649960567436
// MI455X (gfx1250) — compile-verified, matrix operand order checked by reading, not run on hardware
//
#include <hip/hip_runtime.h>
#include <stdint.h>

typedef _Float16 half_t;
typedef __attribute__((ext_vector_type(16))) _Float16 v16h;
typedef __attribute__((ext_vector_type(8)))  float    v8f;

#define T_STEPS 10
#define FEAT 48
#define KCODES 512
#define JOINTD 97
#define NB 8
#define HW 9216            // 96*96
#define NPIX 73728         // 8*9216
#define JLD 128

static const long NTROW = 737280;   // NPIX * T_STEPS

union FragH { v16h v; uint32_t u[8]; _Float16 h[16]; };

__device__ __forceinline__ int kpat(int v){ return (v < 4) ? 2*v : 2*v + 8; }
__device__ __forceinline__ float silu_f(float x){
    return x * __builtin_amdgcn_rcpf(1.0f + __expf(-x));
}

// ---------------------------------------------------------------------------
// Repack B [K,N] (f16, row-major) into WMMA-fragment order:
// Bp[((nt*(K/32)+ks)*32 + lane)*16 + j], j=2v+p -> element (k,n) with
// k = ks*32 + kpat(v) + 8*(lane>>4) + p, n = nt*16 + (lane&15)
// ---------------------------------------------------------------------------
__global__ void k_pack_b(const half_t* __restrict__ Bs, half_t* __restrict__ Bd,
                         int K, int N)
{
    int i = blockIdx.x * blockDim.x + threadIdx.x;
    if (i >= K * N) return;
    int j    = i & 15;
    int lane = (i >> 4) & 31;
    int t    = i >> 9;
    int ktiles = K / 32;
    int nt = t / ktiles, ks = t % ktiles;
    int v = j >> 1, p = j & 1;
    int k = ks * 32 + kpat(v) + 8 * (lane >> 4) + p;
    int n = nt * 16 + (lane & 15);
    Bd[i] = Bs[(long)k * N + n];
}

// ---------------------------------------------------------------------------
// WMMA GEMM: C[M,N] = A[M,K](f16) * Bp(packed f16) + bias (+silu) (+resid)
// grid = (M/128, N/(16*NTILES)), block = 256 (8 waves).
// Each wave computes 16 rows x (16*NTILES) cols, reusing A frags NTILES times.
// Packed B region for the block's n-tiles is contiguous and staged to LDS
// via async global->LDS copies (ASYNCcnt).
// ---------------------------------------------------------------------------
template<int NTILES, int ACT, int RES>
__global__ __launch_bounds__(256)
void k_gemm(const half_t* __restrict__ A, const half_t* __restrict__ Bp,
            const float* __restrict__ bias, const float* __restrict__ resid,
            float* __restrict__ C, long M, int K, int ldc)
{
    extern __shared__ char smem_raw[];
    half_t* ldsB = (half_t*)smem_raw;

    int ktiles = K >> 5;
    // ---- stage packed B tiles (NTILES*ktiles*1024 bytes) into LDS ----
    {
        const char* gsrc =
            (const char*)(Bp + (size_t)blockIdx.y * NTILES * ktiles * 512);
        int nbytes = NTILES * ktiles * 1024;
        uint32_t lbase = (uint32_t)(uintptr_t)ldsB;
        for (int off = threadIdx.x * 16; off < nbytes; off += 256 * 16) {
            uint32_t laddr = lbase + off;
            const char* ga = gsrc + off;
            asm volatile("global_load_async_to_lds_b128 %0, %1, off"
                         :: "v"(laddr), "v"(ga) : "memory");
        }
        asm volatile("s_wait_asynccnt 0x0" ::: "memory");
        __syncthreads();
    }

    int wave = threadIdx.x >> 5;
    int lane = threadIdx.x & 31;
    int lh   = lane >> 4;
    int lc   = lane & 15;
    long m0  = ((long)blockIdx.x * 8 + wave) * 16;
    int n0   = blockIdx.y * NTILES * 16;

    v8f acc[NTILES];
#pragma unroll
    for (int t = 0; t < NTILES; ++t) acc[t] = (v8f){};

    const half_t* Arow = A + (m0 + lc) * (long)K;
    for (int ks = 0; ks < ktiles; ++ks) {
        FragH a;
#pragma unroll
        for (int v = 0; v < 8; ++v)
            a.u[v] = *(const uint32_t*)(Arow + ks * 32 + kpat(v) + 8 * lh);
#pragma unroll
        for (int t = 0; t < NTILES; ++t) {
            FragH b;
            const uint32_t* bw =
                (const uint32_t*)(ldsB + ((t * ktiles + ks) * 32 + lane) * 16);
#pragma unroll
            for (int v = 0; v < 8; ++v) b.u[v] = bw[v];
            acc[t] = __builtin_amdgcn_wmma_f32_16x16x32_f16(
                         false, a.v, false, b.v, (short)0, acc[t], false, false);
        }
    }
#pragma unroll
    for (int t = 0; t < NTILES; ++t) {
        float bval = bias[n0 + t * 16 + lc];
#pragma unroll
        for (int r = 0; r < 8; ++r) {
            long row = m0 + r + 8 * lh;
            long idx = row * (long)ldc + n0 + t * 16 + lc;
            float v = acc[t][r] + bval;
            if (ACT) v = silu_f(v);
            if (RES) v += resid[idx];
            C[idx] = v;
        }
    }
}

// ---------------------------------------------------------------------------
// VQ: WMMA distances + argmin + code/quantized-gather + loss accumulation
// grid = NPIX/128, block = 256. EbTp is codebook^T packed in fragment order.
// ---------------------------------------------------------------------------
__global__ __launch_bounds__(256)
void k_vq(const half_t* __restrict__ J16,   // [NPIX,128]
          const half_t* __restrict__ EbTp,  // packed [128,512]
          const half_t* __restrict__ E16,   // [512,128]
          const float*  __restrict__ Jf,    // [NPIX,128]
          const float*  __restrict__ emb,   // [512,97]
          const float*  __restrict__ embn,  // [512]
          half_t* __restrict__ Q16,         // [NPIX,128]
          float*  __restrict__ codes_out,   // [NPIX]
          float*  __restrict__ loss_out)
{
    int wave = threadIdx.x >> 5;
    int lane = threadIdx.x & 31;
    int lh   = lane >> 4;
    int lc   = lane & 15;
    long m0  = ((long)blockIdx.x * 8 + wave) * 16;

    const half_t* Arow = J16 + (m0 + lc) * (long)JLD;
    FragH a[4];
#pragma unroll
    for (int ks = 0; ks < 4; ++ks)
#pragma unroll
        for (int v = 0; v < 8; ++v)
            a[ks].u[v] = *(const uint32_t*)(Arow + ks * 32 + kpat(v) + 8 * lh);

    float best[8]; int bidx[8];
#pragma unroll
    for (int r = 0; r < 8; ++r) { best[r] = 3.4e38f; bidx[r] = 0; }

    for (int ct = 0; ct < KCODES / 16; ++ct) {
        v8f acc = {};
#pragma unroll
        for (int ks = 0; ks < 4; ++ks) {
            FragH b;
            const uint32_t* bw =
                (const uint32_t*)(EbTp + (((long)ct * 4 + ks) * 32 + lane) * 16);
#pragma unroll
            for (int v = 0; v < 8; ++v) b.u[v] = bw[v];
            acc = __builtin_amdgcn_wmma_f32_16x16x32_f16(
                      false, a[ks].v, false, b.v, (short)0, acc, false, false);
        }
        int code = ct * 16 + lc;
        float en = embn[code];
#pragma unroll
        for (int r = 0; r < 8; ++r) {
            float val = en - 2.0f * acc[r];
            if (val < best[r]) { best[r] = val; bidx[r] = code; }
        }
    }
    // argmin across the 16 lanes of each half (same rows, different codes)
    for (int off = 1; off < 16; off <<= 1) {
#pragma unroll
        for (int r = 0; r < 8; ++r) {
            float ov = __shfl_xor(best[r], off);
            int   oi = __shfl_xor(bidx[r], off);
            if (ov < best[r] || (ov == best[r] && oi < bidx[r])) {
                best[r] = ov; bidx[r] = oi;
            }
        }
    }
    // lanes 0..7 write rows m0..m0+7 ; lanes 16..23 write rows m0+8..m0+15
    int r = lane & 7;
    bool writer = ((lane & 8) == 0) && (lc < 8);
    if (writer) {
        long row = m0 + r + 8 * lh;
        int code = bidx[r];
        codes_out[row] = (float)code;
        const uint4* es = (const uint4*)(E16 + (long)code * JLD);
        uint4* qd = (uint4*)(Q16 + row * (long)JLD);
#pragma unroll
        for (int k = 0; k < JLD / 8; ++k) qd[k] = es[k];
        const float* er = emb + (long)code * JOINTD;
        const float* jr = Jf + row * (long)JLD;
        float s = 0.0f;
        for (int k = 0; k < JOINTD; ++k) { float d = er[k] - jr[k]; s += d * d; }
        atomicAdd(loss_out, s * (1.25f / ((float)NPIX * (float)JOINTD)));
    }
}

// ---------------------------------------------------------------------------
// Fused feature-embed + temporal-proj weights and z0 computation
// ---------------------------------------------------------------------------
__global__ void k_fuse(const float* tp_w, const float* tp_b,
                       const float* fe_w, const float* fe_b,
                       float* Wc, float* bc)
{
    int i = threadIdx.x;              // 0..511
    int o = i >> 3, c = i & 7;
    float s = 0.f;
    for (int f = 0; f < FEAT; ++f) s += tp_w[o * FEAT + f] * fe_w[f * 8 + c];
    Wc[i] = s;
    if (c == 0) {
        float b = tp_b[o];
        for (int f = 0; f < FEAT; ++f) b += tp_w[o * FEAT + f] * fe_b[f];
        bc[o] = b;
    }
}

__global__ void k_embed(const float* __restrict__ x, const float* __restrict__ Wc,
                        const float* __restrict__ bc, float* __restrict__ Z, long total)
{
    long i = (long)blockIdx.x * blockDim.x + threadIdx.x;
    if (i >= total) return;
    int o = i & 63;
    long r = i >> 6;
    int t = (int)(r % T_STEPS);
    long n = r / T_STEPS;
    int b = (int)(n / HW), p = (int)(n % HW);
    const float* xp = x + ((long)b * 80 + t * 8) * HW + p;
    float acc = bc[o];
#pragma unroll
    for (int c = 0; c < 8; ++c) acc += Wc[o * 8 + c] * xp[(long)c * HW];
    Z[i] = acc;
}

// ---------------------------------------------------------------------------
// LayerNorm over channels (C<=64) + SiLU, output f16. one wave per row.
// ---------------------------------------------------------------------------
__global__ __launch_bounds__(256)
void k_ln_silu(const float* __restrict__ X, const float* __restrict__ g,
               const float* __restrict__ gb, half_t* __restrict__ Y16, long M, int C)
{
    int wave = threadIdx.x >> 5, lane = threadIdx.x & 31;
    long row = (long)blockIdx.x * 8 + wave;
    if (row >= M) return;
    const float* x = X + row * (long)C;
    float v0 = (lane < C) ? x[lane] : 0.f;
    float v1 = (lane + 32 < C) ? x[lane + 32] : 0.f;
    float s = v0 + v1;
    for (int o = 16; o > 0; o >>= 1) s += __shfl_xor(s, o);
    float mean = s / (float)C;
    float d0 = (lane < C) ? v0 - mean : 0.f;
    float d1 = (lane + 32 < C) ? v1 - mean : 0.f;
    float q = d0 * d0 + d1 * d1;
    for (int o = 16; o > 0; o >>= 1) q += __shfl_xor(q, o);
    float rs = rsqrtf(q / (float)C + 1e-5f);
    if (lane < C) {
        float y = d0 * rs * g[lane] + gb[lane];
        Y16[row * C + lane] = (half_t)silu_f(y);
    }
    if (lane + 32 < C) {
        int c = lane + 32;
        float y = d1 * rs * g[c] + gb[c];
        Y16[row * C + c] = (half_t)silu_f(y);
    }
}

// ---------------------------------------------------------------------------
// GroupNorm (8 groups of 8ch over [NPIX,64] channel-last) + SiLU
// ---------------------------------------------------------------------------
__global__ __launch_bounds__(256)
void k_gn_stats(const float* __restrict__ X, float* mu, float* rstd)
{
    int bg = blockIdx.x;          // 64 = B*G
    int b = bg >> 3, g = bg & 7;
    float s = 0.f, s2 = 0.f;
    for (int i = threadIdx.x; i < HW * 8; i += blockDim.x) {
        int p = i >> 3, c = i & 7;
        float v = X[((long)b * HW + p) * 64 + g * 8 + c];
        s += v; s2 += v * v;
    }
    __shared__ float sh[512];
    sh[threadIdx.x] = s; sh[256 + threadIdx.x] = s2;
    __syncthreads();
    for (int st = 128; st > 0; st >>= 1) {
        if ((int)threadIdx.x < st) {
            sh[threadIdx.x] += sh[threadIdx.x + st];
            sh[256 + threadIdx.x] += sh[256 + threadIdx.x + st];
        }
        __syncthreads();
    }
    if (threadIdx.x == 0) {
        float inv = 1.0f / ((float)HW * 8.0f);
        float m = sh[0] * inv;
        float var = sh[256] * inv - m * m;
        mu[bg] = m; rstd[bg] = rsqrtf(var + 1e-5f);
    }
}

__global__ void k_gn_apply(const float* __restrict__ X, const float* mu, const float* rstd,
                           const float* __restrict__ g, const float* __restrict__ gb,
                           float* __restrict__ Y, half_t* __restrict__ Y16)
{
    long i = (long)blockIdx.x * blockDim.x + threadIdx.x;
    if (i >= (long)NPIX * 64) return;
    int c = i & 63;
    long n = i >> 6;
    int b = (int)(n / HW);
    int bg = b * 8 + (c >> 3);
    float v = (X[i] - mu[bg]) * rstd[bg] * g[c] + gb[c];
    v = silu_f(v);
    Y[i] = v; Y16[i] = (half_t)v;
}

// ---------------------------------------------------------------------------
// im2col builders (f16 panels)
// ---------------------------------------------------------------------------
__global__ void k_im2col1d(const half_t* __restrict__ S, half_t* __restrict__ A,
                           long M, int C, int Kp, int dil)
{
    long i = (long)blockIdx.x * blockDim.x + threadIdx.x;
    if (i >= M * (long)Kp) return;
    int kp = (int)(i % Kp);
    long r = i / Kp;
    half_t v = (half_t)0.f;
    if (kp < 3 * C) {
        int dt = kp / C, c = kp % C;
        int t = (int)(r % T_STEPS);
        long n = r / T_STEPS;
        int tt = t + dil * (dt - 1);
        if (tt >= 0 && tt < T_STEPS) v = S[(n * T_STEPS + tt) * (long)C + c];
    }
    A[i] = v;
}

__global__ void k_im2col_agg(const float* __restrict__ Z, half_t* __restrict__ A)
{
    long i = (long)blockIdx.x * blockDim.x + threadIdx.x;
    if (i >= (long)NPIX * 640) return;
    int k = (int)(i % 640);
    long n = i / 640;
    int c = k / T_STEPS, t = k % T_STEPS;
    A[i] = (half_t)Z[(n * T_STEPS + t) * 64 + c];
}

__global__ void k_im2col2d(const half_t* __restrict__ S, half_t* __restrict__ A,
                           int C, int Kp, int dil)
{
    long i = (long)blockIdx.x * blockDim.x + threadIdx.x;
    if (i >= (long)NPIX * Kp) return;
    int kp = (int)(i % Kp);
    long n = i / Kp;
    half_t v = (half_t)0.f;
    if (kp < 9 * C) {
        int s = kp / C, c = kp % C;
        int ky = s / 3, kx = s % 3;
        int b = (int)(n / HW), p = (int)(n % HW);
        int h = p / 96, w = p % 96;
        int hh = h + dil * (ky - 1), ww = w + dil * (kx - 1);
        if (hh >= 0 && hh < 96 && ww >= 0 && ww < 96)
            v = S[((long)b * HW + hh * 96 + ww) * (long)C + c];
    }
    A[i] = v;
}

// ---------------------------------------------------------------------------
// Small data-shuffle / convert kernels
// ---------------------------------------------------------------------------
__global__ void k_cvt(const float* s, half_t* d, long n) {
    long i = (long)blockIdx.x * blockDim.x + threadIdx.x;
    if (i < n) d[i] = (half_t)s[i];
}
__global__ void k_zero(float* p, long n) {
    long i = (long)blockIdx.x * blockDim.x + threadIdx.x;
    if (i < n) p[i] = 0.f;
}
__global__ void k_slice_cvt(const float* S, half_t* D, long M, int ld, int off, int C) {
    long i = (long)blockIdx.x * blockDim.x + threadIdx.x;
    if (i >= M * (long)C) return;
    int c = (int)(i % C); long r = i / C;
    D[i] = (half_t)S[r * (long)ld + off + c];
}
__global__ void k_pad_cols(const half_t* S, half_t* D, long M, int C, int Kp) {
    long i = (long)blockIdx.x * blockDim.x + threadIdx.x;
    if (i >= M * (long)Kp) return;
    int kp = (int)(i % Kp); long r = i / Kp;
    D[i] = (kp < C) ? S[r * (long)C + kp] : (half_t)0.f;
}
__global__ void k_add(const float* a, const float* b, float* y, half_t* y16, long n) {
    long i = (long)blockIdx.x * blockDim.x + threadIdx.x;
    if (i >= n) return;
    float v = a[i] + b[i];
    y[i] = v; y16[i] = (half_t)v;
}
__global__ void k_build_joint(const float* __restrict__ TA, float* __restrict__ Jf) {
    long i = (long)blockIdx.x * blockDim.x + threadIdx.x;
    if (i >= (long)NPIX * JLD) return;
    int c = (int)(i & (JLD - 1));
    long n = i >> 7;
    if (c >= 48 && c < 96) return;      // ctx filled later by cp GEMM
    float v = 0.f;
    if (c < 48) v = TA[n * 64 + c];
    else if (c == 96) v = TA[n * 64 + 48];   // AGE_RESCALE == 1.0
    Jf[i] = v;
}

// ---------------------------------------------------------------------------
// Weight preparation (f32 -> f16 B matrices [K,N], K padded)
// ---------------------------------------------------------------------------
__global__ void k_prep_dense(const float* W, half_t* Bm, int O, int I, int Kp) {
    int i = blockIdx.x * blockDim.x + threadIdx.x;
    if (i >= Kp * O) return;
    int o = i % O, k = i / O;
    Bm[i] = (k < I) ? (half_t)W[o * I + k] : (half_t)0.f;
}
__global__ void k_prep_conv1d(const float* W, half_t* Bm, int O, int I, int Kp) {
    int i = blockIdx.x * blockDim.x + threadIdx.x;
    if (i >= Kp * O) return;
    int o = i % O, k = i / O;
    half_t v = (half_t)0.f;
    if (k < 3 * I) v = (half_t)W[o * I * 3 + (k % I) * 3 + (k / I)];
    Bm[i] = v;
}
__global__ void k_prep_conv2d(const float* W, half_t* Bm, int O, int CI, int Kp) {
    int i = blockIdx.x * blockDim.x + threadIdx.x;
    if (i >= Kp * O) return;
    int o = i % O, k = i / O;
    half_t v = (half_t)0.f;
    if (k < 9 * CI) { int s = k / CI, c = k % CI; v = (half_t)W[o * CI * 9 + c * 9 + s]; }
    Bm[i] = v;
}
__global__ void k_prep_trajage(const float* trajW, const float* ageW, half_t* Bm) {
    int i = blockIdx.x * blockDim.x + threadIdx.x;
    if (i >= 64 * 64) return;
    int o = i % 64, k = i / 64;
    float v = 0.f;
    if (o < 48) v = trajW[o * 64 + k];
    else if (o == 48) v = ageW[k];
    Bm[i] = (half_t)v;
}
__global__ void k_prep_tabias(const float* tb, const float* ab, float* bias) {
    int i = threadIdx.x;
    if (i < 64) bias[i] = (i < 48) ? tb[i] : (i == 48 ? ab[0] : 0.f);
}
__global__ void k_prep_codebook(const float* emb, half_t* EbT, half_t* E16, float* embn) {
    int i = blockIdx.x * blockDim.x + threadIdx.x;
    if (i >= KCODES * JLD) return;
    int k = i % JLD, j = i / JLD;
    float v = (k < JOINTD) ? emb[j * JOINTD + k] : 0.f;
    E16[i] = (half_t)v;
    EbT[k * KCODES + j] = (half_t)v;
    if (k == 0) {
        float s = 0.f;
        for (int c = 0; c < JOINTD; ++c) { float e = emb[j * JOINTD + c]; s += e * e; }
        embn[j] = s;
    }
}

// ---------------------------------------------------------------------------
// Final feature-decode: recon[b, t*8+o, h, w]
// ---------------------------------------------------------------------------
__global__ void k_fd(const float* __restrict__ Zd, const float* __restrict__ fw,
                     const float* __restrict__ fb, float* __restrict__ out)
{
    long i = (long)blockIdx.x * blockDim.x + threadIdx.x;
    if (i >= (long)737280 * 8) return;
    int o = (int)(i & 7);
    long r = i >> 3;
    int t = (int)(r % T_STEPS);
    long n = r / T_STEPS;
    int b = (int)(n / HW), p = (int)(n % HW);
    float acc = fb[o];
    const float* zr = Zd + r * FEAT;
#pragma unroll 8
    for (int f = 0; f < FEAT; ++f) acc += fw[o * FEAT + f] * zr[f];
    out[(long)b * 80 * HW + (long)(t * 8 + o) * HW + p] = acc;
}

// ===========================================================================
extern "C" void kernel_launch(void* const* d_in, const int* in_sizes, int n_in,
                              void* d_out, int out_size, void* d_ws, size_t ws_size,
                              hipStream_t stream)
{
    (void)in_sizes; (void)n_in; (void)out_size; (void)ws_size;
    const long NT = NTROW;
    const long NPx = NPIX;

    auto F = [&](int i) { return (const float*)d_in[i]; };
    // input indices (setup_inputs dict insertion order, nested dicts in order)
    const float* x      = F(0);
    const float* fe_w   = F(1);  const float* fe_b = F(2);
    const float* tp_w   = F(3);  const float* tp_b = F(4);
    // tb1: 5..12, tb2: 13..20
    const float* agg_b  = F(22);
    const float* traj_w = F(23); const float* traj_b = F(24);
    const float* age_w  = F(25); const float* age_b  = F(26);
    const float* cp_b   = F(62);
    const float* cbook  = F(63);
    const float* c2t_b  = F(65);
    const float* td_b   = F(67);
    const float* fd_w   = F(76); const float* fd_b = F(77);

    // ---- workspace bump allocator -------------------------------------
    char* wptr = (char*)d_ws;
    auto alloc = [&](size_t bytes) -> void* {
        void* p = (void*)wptr;
        wptr += (bytes + 255) & ~(size_t)255;
        return p;
    };
    float*  Z     = (float*)alloc(NT * 64 * 4);
    float*  Hbuf  = (float*)alloc(NT * 64 * 4);          // also reused as Hd [NT,48]
    half_t* L16   = (half_t*)alloc(NT * 64 * 2);         // also reused as Ld16 [NT,48]
    half_t* A16   = (half_t*)alloc(NT * 192 * 2);        // largest im2col panel
    float*  zagg  = (float*)alloc(NPx * 64 * 4);
    half_t* zagg16= (half_t*)alloc(NPx * 64 * 2);
    float*  TA    = (float*)alloc(NPx * 64 * 4);
    float*  Jf    = (float*)alloc(NPx * JLD * 4);
    half_t* J16   = (half_t*)alloc(NPx * JLD * 2);
    half_t* s0_16 = (half_t*)alloc(NPx * 48 * 2);
    half_t* s0p16 = (half_t*)alloc(NPx * 64 * 2);
    float*  t2    = (float*)alloc(NPx * 64 * 4);
    float*  t3    = (float*)alloc(NPx * 64 * 4);
    half_t* t3h   = (half_t*)alloc(NPx * 64 * 2);
    float*  skipb = (float*)alloc(NPx * 64 * 4);
    float*  sbuf  = (float*)alloc(NPx * 64 * 4);
    half_t* s16   = (half_t*)alloc(NPx * 64 * 2);
    float*  mu    = (float*)alloc(64 * 4);
    float*  rstd  = (float*)alloc(64 * 4);
    half_t* Q16   = (half_t*)alloc(NPx * JLD * 2);
    float*  d1    = (float*)alloc(NPx * 64 * 4);
    half_t* d1h   = (half_t*)alloc(NPx * 64 * 2);
    float*  Zd    = (float*)alloc(NT * 48 * 4);
    float*  Wc    = (float*)alloc(64 * 8 * 4);
    float*  bc    = (float*)alloc(64 * 4);
    // raw (unpacked) B matrices
    half_t* Bt[4]; for (int i = 0; i < 4; ++i) Bt[i] = (half_t*)alloc(192 * 64 * 2);
    half_t* Bd1   = (half_t*)alloc(160 * 48 * 2);
    half_t* Bd2   = (half_t*)alloc(160 * 48 * 2);
    half_t* Bagg  = (half_t*)alloc(640 * 64 * 2);
    half_t* Bta   = (half_t*)alloc(64 * 64 * 2);
    float*  biasTA= (float*)alloc(64 * 4);
    half_t* Bs1a  = (half_t*)alloc(448 * 64 * 2);
    half_t* Bsb[7]; for (int i = 0; i < 7; ++i) Bsb[i] = (half_t*)alloc(576 * 64 * 2);
    half_t* Bskip = (half_t*)alloc(64 * 64 * 2);
    half_t* Bcp   = (half_t*)alloc(64 * 48 * 2);
    half_t* EbT   = (half_t*)alloc(JLD * KCODES * 2);
    half_t* E16   = (half_t*)alloc(KCODES * JLD * 2);
    float*  embn  = (float*)alloc(KCODES * 4);
    half_t* Bc2t  = (half_t*)alloc(128 * 64 * 2);
    half_t* Btd   = (half_t*)alloc(64 * 480 * 2);
    // packed (fragment-order) B matrices
    half_t* BtP[4]; for (int i = 0; i < 4; ++i) BtP[i] = (half_t*)alloc(192 * 64 * 2);
    half_t* Bd1P  = (half_t*)alloc(160 * 48 * 2);
    half_t* Bd2P  = (half_t*)alloc(160 * 48 * 2);
    half_t* BaggP = (half_t*)alloc(640 * 64 * 2);
    half_t* BtaP  = (half_t*)alloc(64 * 64 * 2);
    half_t* Bs1aP = (half_t*)alloc(448 * 64 * 2);
    half_t* BsbP[7]; for (int i = 0; i < 7; ++i) BsbP[i] = (half_t*)alloc(576 * 64 * 2);
    half_t* BskipP= (half_t*)alloc(64 * 64 * 2);
    half_t* BcpP  = (half_t*)alloc(64 * 48 * 2);
    half_t* EbTP  = (half_t*)alloc(JLD * KCODES * 2);
    half_t* Bc2tP = (half_t*)alloc(128 * 64 * 2);
    half_t* BtdP  = (half_t*)alloc(64 * 480 * 2);

    float* recon     = (float*)d_out;
    float* loss_out  = (float*)d_out + (long)NB * 80 * HW;
    float* codes_out = loss_out + 1;

    auto g1 = [](long n) { return (unsigned)((n + 255) / 256); };
    auto shb = [](int NTILES, int K) { return (size_t)NTILES * (K / 32) * 1024; };
    dim3 blk(256);

    // ---- weight prep --------------------------------------------------
    k_fuse<<<1, 512, 0, stream>>>(tp_w, tp_b, fe_w, fe_b, Wc, bc);
    k_prep_conv1d<<<g1(192 * 64), blk, 0, stream>>>(F(7),  Bt[0], 64, 64, 192);
    k_prep_conv1d<<<g1(192 * 64), blk, 0, stream>>>(F(11), Bt[1], 64, 64, 192);
    k_prep_conv1d<<<g1(192 * 64), blk, 0, stream>>>(F(15), Bt[2], 64, 64, 192);
    k_prep_conv1d<<<g1(192 * 64), blk, 0, stream>>>(F(19), Bt[3], 64, 64, 192);
    k_prep_conv1d<<<g1(160 * 48), blk, 0, stream>>>(F(70), Bd1, 48, 48, 160);
    k_prep_conv1d<<<g1(160 * 48), blk, 0, stream>>>(F(74), Bd2, 48, 48, 160);
    k_prep_dense<<<g1(640 * 64), blk, 0, stream>>>(F(21), Bagg, 64, 640, 640);
    k_prep_trajage<<<g1(64 * 64), blk, 0, stream>>>(traj_w, age_w, Bta);
    k_prep_tabias<<<1, 64, 0, stream>>>(traj_b, age_b, biasTA);
    k_prep_conv2d<<<g1(448 * 64), blk, 0, stream>>>(F(27), Bs1a, 64, 48, 448);
    k_prep_conv2d<<<g1(576 * 64), blk, 0, stream>>>(F(31), Bsb[0], 64, 64, 576);
    k_prep_conv2d<<<g1(576 * 64), blk, 0, stream>>>(F(37), Bsb[1], 64, 64, 576);
    k_prep_conv2d<<<g1(576 * 64), blk, 0, stream>>>(F(41), Bsb[2], 64, 64, 576);
    k_prep_conv2d<<<g1(576 * 64), blk, 0, stream>>>(F(45), Bsb[3], 64, 64, 576);
    k_prep_conv2d<<<g1(576 * 64), blk, 0, stream>>>(F(49), Bsb[4], 64, 64, 576);
    k_prep_conv2d<<<g1(576 * 64), blk, 0, stream>>>(F(53), Bsb[5], 64, 64, 576);
    k_prep_conv2d<<<g1(576 * 64), blk, 0, stream>>>(F(57), Bsb[6], 64, 64, 576);
    k_prep_dense<<<g1(64 * 64), blk, 0, stream>>>(F(35), Bskip, 64, 48, 64);
    k_prep_dense<<<g1(64 * 48), blk, 0, stream>>>(F(61), Bcp, 48, 64, 64);
    k_prep_codebook<<<g1(KCODES * JLD), blk, 0, stream>>>(cbook, EbT, E16, embn);
    k_prep_dense<<<g1(128 * 64), blk, 0, stream>>>(F(64), Bc2t, 64, 97, 128);
    k_prep_dense<<<g1(64 * 480), blk, 0, stream>>>(F(66), Btd, 480, 64, 64);
    // pack everything into fragment order
    for (int i = 0; i < 4; ++i)
        k_pack_b<<<g1(192 * 64), blk, 0, stream>>>(Bt[i], BtP[i], 192, 64);
    k_pack_b<<<g1(160 * 48), blk, 0, stream>>>(Bd1, Bd1P, 160, 48);
    k_pack_b<<<g1(160 * 48), blk, 0, stream>>>(Bd2, Bd2P, 160, 48);
    k_pack_b<<<g1(640 * 64), blk, 0, stream>>>(Bagg, BaggP, 640, 64);
    k_pack_b<<<g1(64 * 64), blk, 0, stream>>>(Bta, BtaP, 64, 64);
    k_pack_b<<<g1(448 * 64), blk, 0, stream>>>(Bs1a, Bs1aP, 448, 64);
    for (int i = 0; i < 7; ++i)
        k_pack_b<<<g1(576 * 64), blk, 0, stream>>>(Bsb[i], BsbP[i], 576, 64);
    k_pack_b<<<g1(64 * 64), blk, 0, stream>>>(Bskip, BskipP, 64, 64);
    k_pack_b<<<g1(64 * 48), blk, 0, stream>>>(Bcp, BcpP, 64, 48);
    k_pack_b<<<g1(JLD * KCODES), blk, 0, stream>>>(EbT, EbTP, JLD, KCODES);
    k_pack_b<<<g1(128 * 64), blk, 0, stream>>>(Bc2t, Bc2tP, 128, 64);
    k_pack_b<<<g1(64 * 480), blk, 0, stream>>>(Btd, BtdP, 64, 480);

    // ---- temporal encoder --------------------------------------------
    k_embed<<<g1(NT * 64), blk, 0, stream>>>(x, Wc, bc, Z, NT * 64);

    const int tb_base[2] = {5, 13};
    const int tb_dil[2]  = {1, 2};
    for (int t = 0; t < 2; ++t) {
        int bi = tb_base[t], dil = tb_dil[t];
        k_ln_silu<<<g1(NT * 32), blk, 0, stream>>>(Z, F(bi), F(bi + 1), L16, NT, 64);
        k_im2col1d<<<g1(NT * 192), blk, 0, stream>>>(L16, A16, NT, 64, 192, dil);
        k_gemm<4,0,0><<<dim3(NT / 128, 1), blk, shb(4, 192), stream>>>(
            A16, BtP[2 * t], F(bi + 3), nullptr, Hbuf, NT, 192, 64);
        k_ln_silu<<<g1(NT * 32), blk, 0, stream>>>(Hbuf, F(bi + 4), F(bi + 5), L16, NT, 64);
        k_im2col1d<<<g1(NT * 192), blk, 0, stream>>>(L16, A16, NT, 64, 192, dil);
        k_gemm<4,0,1><<<dim3(NT / 128, 1), blk, shb(4, 192), stream>>>(
            A16, BtP[2 * t + 1], F(bi + 7), Z, Z, NT, 192, 64);
    }
    // aggregation + traj/age
    k_im2col_agg<<<g1(NPx * 640), blk, 0, stream>>>(Z, A16);
    k_gemm<4,0,0><<<dim3(NPx / 128, 1), blk, shb(4, 640), stream>>>(
        A16, BaggP, agg_b, nullptr, zagg, NPx, 640, 64);
    k_cvt<<<g1(NPx * 64), blk, 0, stream>>>(zagg, zagg16, NPx * 64);
    k_gemm<4,0,0><<<dim3(NPx / 128, 1), blk, shb(4, 64), stream>>>(
        zagg16, BtaP, biasTA, nullptr, TA, NPx, 64, 64);
    k_build_joint<<<g1(NPx * JLD), blk, 0, stream>>>(TA, Jf);

    // ---- spatial stack -----------------------------------------------
    k_slice_cvt<<<g1(NPx * 48), blk, 0, stream>>>(Jf, s0_16, NPx, JLD, 0, 48);
    // sb1 (dil=1, ci=48, has skip conv)
    k_im2col2d<<<g1(NPx * 448), blk, 0, stream>>>(s0_16, A16, 48, 448, 1);
    k_gemm<4,0,0><<<dim3(NPx / 128, 1), blk, shb(4, 448), stream>>>(
        A16, Bs1aP, F(28), nullptr, t2, NPx, 448, 64);
    k_gn_stats<<<64, blk, 0, stream>>>(t2, mu, rstd);
    k_gn_apply<<<g1(NPx * 64), blk, 0, stream>>>(t2, mu, rstd, F(29), F(30), t3, t3h);
    k_im2col2d<<<g1(NPx * 576), blk, 0, stream>>>(t3h, A16, 64, 576, 1);
    k_gemm<4,0,0><<<dim3(NPx / 128, 1), blk, shb(4, 576), stream>>>(
        A16, BsbP[0], F(32), nullptr, t2, NPx, 576, 64);
    k_gn_stats<<<64, blk, 0, stream>>>(t2, mu, rstd);
    k_gn_apply<<<g1(NPx * 64), blk, 0, stream>>>(t2, mu, rstd, F(33), F(34), t3, t3h);
    k_pad_cols<<<g1(NPx * 64), blk, 0, stream>>>(s0_16, s0p16, NPx, 48, 64);
    k_gemm<4,0,0><<<dim3(NPx / 128, 1), blk, shb(4, 64), stream>>>(
        s0p16, BskipP, F(36), nullptr, skipb, NPx, 64, 64);
    k_add<<<g1(NPx * 64), blk, 0, stream>>>(skipb, t3, sbuf, s16, NPx * 64);
    // sb2..sb4 (identity skip)
    const int sb_base[3] = {37, 45, 53};
    const int sb_dil[3]  = {2, 4, 8};
    for (int sbi = 0; sbi < 3; ++sbi) {
        int bi = sb_base[sbi], dil = sb_dil[sbi];
        k_im2col2d<<<g1(NPx * 576), blk, 0, stream>>>(s16, A16, 64, 576, dil);
        k_gemm<4,0,0><<<dim3(NPx / 128, 1), blk, shb(4, 576), stream>>>(
            A16, BsbP[1 + 2 * sbi], F(bi + 1), nullptr, t2, NPx, 576, 64);
        k_gn_stats<<<64, blk, 0, stream>>>(t2, mu, rstd);
        k_gn_apply<<<g1(NPx * 64), blk, 0, stream>>>(t2, mu, rstd, F(bi + 2), F(bi + 3), t3, t3h);
        k_im2col2d<<<g1(NPx * 576), blk, 0, stream>>>(t3h, A16, 64, 576, dil);
        k_gemm<4,0,0><<<dim3(NPx / 128, 1), blk, shb(4, 576), stream>>>(
            A16, BsbP[2 + 2 * sbi], F(bi + 5), nullptr, t2, NPx, 576, 64);
        k_gn_stats<<<64, blk, 0, stream>>>(t2, mu, rstd);
        k_gn_apply<<<g1(NPx * 64), blk, 0, stream>>>(t2, mu, rstd, F(bi + 6), F(bi + 7), t3, t3h);
        k_add<<<g1(NPx * 64), blk, 0, stream>>>(sbuf, t3, sbuf, s16, NPx * 64);
    }
    // cp -> ctx written directly into joint cols 48..95
    k_gemm<3,0,0><<<dim3(NPx / 128, 1), blk, shb(3, 64), stream>>>(
        s16, BcpP, cp_b, nullptr, Jf + 48, NPx, 64, JLD);

    // ---- VQ ------------------------------------------------------------
    k_cvt<<<g1(NPx * JLD), blk, 0, stream>>>(Jf, J16, NPx * JLD);
    k_zero<<<1, 1, 0, stream>>>(loss_out, 1);
    k_vq<<<NPx / 128, blk, 0, stream>>>(J16, EbTP, E16, Jf, cbook, embn,
                                        Q16, codes_out, loss_out);

    // ---- decoder -------------------------------------------------------
    k_gemm<4,1,0><<<dim3(NPx / 128, 1), blk, shb(4, 128), stream>>>(
        Q16, Bc2tP, c2t_b, nullptr, d1, NPx, 128, 64);
    k_cvt<<<g1(NPx * 64), blk, 0, stream>>>(d1, d1h, NPx * 64);
    k_gemm<5,1,0><<<dim3(NPx / 128, 6), blk, shb(5, 64), stream>>>(
        d1h, BtdP, td_b, nullptr, Zd, NPx, 64, 480);
    // db1 res-block on [NT,48]
    k_ln_silu<<<g1(NT * 32), blk, 0, stream>>>(Zd, F(68), F(69), L16, NT, 48);
    k_im2col1d<<<g1(NT * 160), blk, 0, stream>>>(L16, A16, NT, 48, 160, 1);
    k_gemm<3,0,0><<<dim3(NT / 128, 1), blk, shb(3, 160), stream>>>(
        A16, Bd1P, F(71), nullptr, Hbuf, NT, 160, 48);
    k_ln_silu<<<g1(NT * 32), blk, 0, stream>>>(Hbuf, F(72), F(73), L16, NT, 48);
    k_im2col1d<<<g1(NT * 160), blk, 0, stream>>>(L16, A16, NT, 48, 160, 1);
    k_gemm<3,0,1><<<dim3(NT / 128, 1), blk, shb(3, 160), stream>>>(
        A16, Bd2P, F(75), Zd, Zd, NT, 160, 48);
    // final feature decode -> recon (NCHW)
    k_fd<<<g1(NT * 8), blk, 0, stream>>>(Zd, fd_w, fd_b, recon);
}
